// GatedGCNLayer_20134806684396
// MI455X (gfx1250) — matrix operand order repaired and checked by reading, not compiled, not run on hardware
//
#include <hip/hip_runtime.h>
#include <hip/hip_bf16.h>

#define D_DIM 128

typedef __attribute__((ext_vector_type(16))) _Float16 v16h;
typedef __attribute__((ext_vector_type(8)))  float    v8f;

// ---- WMMA fragment loaders (wave32, 16x16x32 f16) ------------------------

// A 16x32 f16 tile from f32 row-major [rows, 128].
// Lane l<16: M=l, halves 0..7 -> K=kbase..+7, halves 8..15 -> K=kbase+16..+23
// Lane l>=16: same with +8 K offset.
__device__ __forceinline__ v16h load_a_f32(const float* __restrict__ row0ptr,
                                           int kbase, int lane) {
  const int m   = lane & 15;
  const int sel = (lane >> 4) << 3;
  const float* p = row0ptr + (size_t)m * D_DIM + kbase + sel;
  v16h a;
#pragma unroll
  for (int j = 0; j < 8; ++j) a[j] = (_Float16)p[j];
#pragma unroll
  for (int j = 0; j < 8; ++j) a[j + 8] = (_Float16)p[16 + j];
  return a;
}

// B 32x16 f16 tile from pre-transposed f16 weights Wt[out][in] (row-major,
// contiguous K). Lanes 0-15 hold K=kbase..+7,+16..+23; lanes 16-31 same +8.
__device__ __forceinline__ v16h load_b_h(const _Float16* __restrict__ wt,
                                         int n0, int kbase, int lane) {
  const _Float16* p = wt + (size_t)(n0 + (lane & 15)) * D_DIM + kbase + ((lane >> 4) << 3);
  v16h b;
#pragma unroll
  for (int j = 0; j < 16; ++j) b[j] = p[j < 8 ? j : j + 8];
  return b;
}

#define WMMA(a, b, c) \
  __builtin_amdgcn_wmma_f32_16x16x32_f16(false, (a), false, (b), (short)0, (c), false, false)

// ---- Kernel 0: transpose+convert the 8 weight matrices to f16 [out][in] ---
__global__ void __launch_bounds__(256) k_wt(
    const float* __restrict__ w0, const float* __restrict__ w1,
    const float* __restrict__ w2, const float* __restrict__ w3,
    const float* __restrict__ w4, const float* __restrict__ w5,
    const float* __restrict__ w6, const float* __restrict__ w7,
    _Float16* __restrict__ wt) {
  const float* srcs[8] = {w0, w1, w2, w3, w4, w5, w6, w7};
  const float* w = srcs[blockIdx.y];
  _Float16* o = wt + (size_t)blockIdx.y * 16384;
  int idx = blockIdx.x * 256 + threadIdx.x;       // 0..16383, grid.x = 64
  int n = idx >> 7, k = idx & 127;
  o[(size_t)n * D_DIM + k] = (_Float16)w[(size_t)k * D_DIM + n];  // W stored [in,out]
}

// ---- Kernel 1: 5 fused node GEMMs: h_emb, Uh, Vh, W2h, W3h ---------------
__global__ void __launch_bounds__(32) k_node_gemm(
    const float* __restrict__ h, const _Float16* __restrict__ wt,
    const float* __restrict__ Ub, const float* __restrict__ Vb,
    const float* __restrict__ W2b, const float* __restrict__ W3b,
    float* __restrict__ h_emb, float* __restrict__ Uh, float* __restrict__ Vh,
    float* __restrict__ W2h, float* __restrict__ W3h, int N) {
  const int row0 = blockIdx.x * 16;
  if (row0 + 16 > N) return;
  const int lane = threadIdx.x;
  const _Float16* w; const float* bias; float* out;
  switch (blockIdx.y) {
    case 0:  w = wt + 0 * 16384; bias = nullptr; out = h_emb; break;
    case 1:  w = wt + 1 * 16384; bias = Ub;      out = Uh;    break;
    case 2:  w = wt + 2 * 16384; bias = Vb;      out = Vh;    break;
    case 3:  w = wt + 3 * 16384; bias = W2b;     out = W2h;   break;
    default: w = wt + 4 * 16384; bias = W3b;     out = W3h;   break;
  }
  v16h a[4];
#pragma unroll
  for (int kt = 0; kt < 4; ++kt) a[kt] = load_a_f32(h + (size_t)row0 * D_DIM, kt * 32, lane);
  const int m_off = (lane >> 4) << 3;
  const int n = lane & 15;
#pragma unroll
  for (int nt = 0; nt < 8; ++nt) {
    v8f c = {};
#pragma unroll
    for (int kt = 0; kt < 4; ++kt) {
      v16h b = load_b_h(w, nt * 16, kt * 32, lane);
      c = WMMA(a[kt], b, c);
    }
    const int col = nt * 16 + n;
    const float bv = bias ? bias[col] : 0.0f;
#pragma unroll
    for (int r = 0; r < 8; ++r)
      out[(size_t)(row0 + r + m_off) * D_DIM + col] = c[r] + bv;
  }
}

// ---- Kernel 2: edge GEMMs e_emb & W1e fused with src/dst gather and
//      BN statistics accumulation (pre_e = W1e + W2h[src] + W3h[dst]) ------
__global__ void __launch_bounds__(256) k_edge_gemm(
    const float* __restrict__ e, const _Float16* __restrict__ wt,
    const float* __restrict__ W1b, const int* __restrict__ src,
    const int* __restrict__ dst, const float* __restrict__ W2h,
    const float* __restrict__ W3h, float* __restrict__ e_emb,
    float* __restrict__ preE, float* __restrict__ statsE, int tilesE) {
  __shared__ float ls[256];                 // [0..127]=sum, [128..255]=sumsq
  const int tid = threadIdx.x;
  ls[tid] = 0.0f;
  __syncthreads();
  const int lane = tid & 31;
  const int tile = blockIdx.x * 8 + (tid >> 5);
  if (tile < tilesE) {
    const int row0 = tile * 16;
    const _Float16* wE = wt + 5 * 16384;
    const _Float16* w1 = wt + 6 * 16384;
    v16h a[4];
#pragma unroll
    for (int kt = 0; kt < 4; ++kt) a[kt] = load_a_f32(e + (size_t)row0 * D_DIM, kt * 32, lane);
    const int m_off = (lane >> 4) << 3;
    const int n = lane & 15;
#pragma unroll
    for (int nt = 0; nt < 8; ++nt) {
      v8f ce = {};
      v8f c1 = {};
#pragma unroll
      for (int kt = 0; kt < 4; ++kt) {
        v16h b0 = load_b_h(wE, nt * 16, kt * 32, lane);
        ce = WMMA(a[kt], b0, ce);
        v16h b1 = load_b_h(w1, nt * 16, kt * 32, lane);
        c1 = WMMA(a[kt], b1, c1);
      }
      const int col = nt * 16 + n;
      const float bv = W1b[col];
      float s1 = 0.0f, s2 = 0.0f;
#pragma unroll
      for (int r = 0; r < 8; ++r) {
        const int row = row0 + r + m_off;
        const size_t o = (size_t)row * D_DIM + col;
        e_emb[o] = ce[r];
        const int s = src[row], d0 = dst[row];
        const float v = c1[r] + bv + W2h[(size_t)s * D_DIM + col]
                               + W3h[(size_t)d0 * D_DIM + col];
        preE[o] = v;
        s1 += v;
        s2 += v * v;
      }
      atomicAdd(&ls[col], s1);
      atomicAdd(&ls[128 + col], s2);
    }
  }
  __syncthreads();
  atomicAdd(&statsE[tid], ls[tid]);
}

// ---- Kernel: finalize batch-norm stats -> scale/shift (a,b) --------------
__global__ void __launch_bounds__(128) k_bnfin(
    const float* __restrict__ stats, const float* __restrict__ gamma,
    const float* __restrict__ beta, float invN, float* __restrict__ ab) {
  const int c = threadIdx.x;                // 128 threads
  const float m   = stats[c] * invN;
  const float var = stats[128 + c] * invN - m * m;
  const float inv = rsqrtf(var + 1e-5f);
  const float aa  = gamma[c] * inv;
  ab[c]       = aa;
  ab[128 + c] = beta[c] - m * aa;
}

// ---- Kernel: e_new = e_emb + relu(a*pre_e + b) ---------------------------
__global__ void __launch_bounds__(256) k_enew(
    const float* __restrict__ e_emb, const float* __restrict__ preE,
    const float* __restrict__ ab, float* __restrict__ e_new, int total4) {
  const int i = blockIdx.x * 256 + threadIdx.x;
  if (i >= total4) return;
  const int c0 = (i & 31) * 4;
  const float4 p = ((const float4*)preE)[i];
  const float4 g = ((const float4*)e_emb)[i];
  float4 o;
  o.x = g.x + fmaxf(p.x * ab[c0 + 0] + ab[128 + c0 + 0], 0.0f);
  o.y = g.y + fmaxf(p.y * ab[c0 + 1] + ab[128 + c0 + 1], 0.0f);
  o.z = g.z + fmaxf(p.z * ab[c0 + 2] + ab[128 + c0 + 2], 0.0f);
  o.w = g.w + fmaxf(p.w * ab[c0 + 3] + ab[128 + c0 + 3], 0.0f);
  ((float4*)e_new)[i] = o;
}

// ---- Kernel: sigma = sigmoid(e_new) @ W_eta, fused scatter to num/den ----
__global__ void __launch_bounds__(32) k_sigma(
    const float* __restrict__ e_new, const _Float16* __restrict__ w_eta,
    const int* __restrict__ src, const int* __restrict__ dst,
    const float* __restrict__ Vh, float* __restrict__ num,
    float* __restrict__ den, int tilesE) {
  const int tile = blockIdx.x;
  if (tile >= tilesE) return;
  const int lane = threadIdx.x;
  const int row0 = tile * 16;
  const int m   = lane & 15;
  const int sel = (lane >> 4) << 3;
  v16h a[4];
#pragma unroll
  for (int kt = 0; kt < 4; ++kt) {
    const float* p = e_new + (size_t)(row0 + m) * D_DIM + kt * 32 + sel;
#pragma unroll
    for (int j = 0; j < 8; ++j)
      a[kt][j]     = (_Float16)(1.0f / (1.0f + __expf(-p[j])));
#pragma unroll
    for (int j = 0; j < 8; ++j)
      a[kt][j + 8] = (_Float16)(1.0f / (1.0f + __expf(-p[16 + j])));
  }
  const int m_off = (lane >> 4) << 3;
  const int n = lane & 15;
#pragma unroll
  for (int nt = 0; nt < 8; ++nt) {
    v8f c = {};
#pragma unroll
    for (int kt = 0; kt < 4; ++kt) {
      v16h b = load_b_h(w_eta, nt * 16, kt * 32, lane);
      c = WMMA(a[kt], b, c);
    }
    const int col = nt * 16 + n;
#pragma unroll
    for (int r = 0; r < 8; ++r) {
      const int row = row0 + r + m_off;
      const int s = src[row], d0 = dst[row];
      const float sg = c[r];
      const float v  = Vh[(size_t)s * D_DIM + col];
      atomicAdd(&num[(size_t)d0 * D_DIM + col], sg * v);
      atomicAdd(&den[(size_t)d0 * D_DIM + col], sg);
    }
  }
}

// ---- Kernel: x = Uh + num/(den+eps), accumulate node BN stats ------------
__global__ void __launch_bounds__(256) k_agg(
    const float* __restrict__ Uh, const float* __restrict__ num,
    const float* __restrict__ den, float* __restrict__ x,
    float* __restrict__ statsH, int N) {
  __shared__ float ls[256];
  const int tid = threadIdx.x;
  ls[tid] = 0.0f;
  __syncthreads();
  const int row = blockIdx.x * 8 + (tid >> 5);
  const int c0 = (tid & 31) * 4;
  if (row < N) {
    const size_t o = (size_t)row * D_DIM + c0;
    const float4 u  = *(const float4*)(Uh + o);
    const float4 nm = *(const float4*)(num + o);
    const float4 dn = *(const float4*)(den + o);
    float4 xv;
    xv.x = u.x + nm.x / (dn.x + 1e-5f);
    xv.y = u.y + nm.y / (dn.y + 1e-5f);
    xv.z = u.z + nm.z / (dn.z + 1e-5f);
    xv.w = u.w + nm.w / (dn.w + 1e-5f);
    *(float4*)(x + o) = xv;
    atomicAdd(&ls[c0 + 0], xv.x); atomicAdd(&ls[128 + c0 + 0], xv.x * xv.x);
    atomicAdd(&ls[c0 + 1], xv.y); atomicAdd(&ls[128 + c0 + 1], xv.y * xv.y);
    atomicAdd(&ls[c0 + 2], xv.z); atomicAdd(&ls[128 + c0 + 2], xv.z * xv.z);
    atomicAdd(&ls[c0 + 3], xv.w); atomicAdd(&ls[128 + c0 + 3], xv.w * xv.w);
  }
  __syncthreads();
  atomicAdd(&statsH[tid], ls[tid]);
}

// ---- Kernel: h_new = h_emb + relu(a*x + b) -------------------------------
__global__ void __launch_bounds__(256) k_hnew(
    const float* __restrict__ h_emb, const float* __restrict__ x,
    const float* __restrict__ ab, float* __restrict__ h_new, int total4) {
  const int i = blockIdx.x * 256 + threadIdx.x;
  if (i >= total4) return;
  const int c0 = (i & 31) * 4;
  const float4 p = ((const float4*)x)[i];
  const float4 g = ((const float4*)h_emb)[i];
  float4 o;
  o.x = g.x + fmaxf(p.x * ab[c0 + 0] + ab[128 + c0 + 0], 0.0f);
  o.y = g.y + fmaxf(p.y * ab[c0 + 1] + ab[128 + c0 + 1], 0.0f);
  o.z = g.z + fmaxf(p.z * ab[c0 + 2] + ab[128 + c0 + 2], 0.0f);
  o.w = g.w + fmaxf(p.w * ab[c0 + 3] + ab[128 + c0 + 3], 0.0f);
  ((float4*)h_new)[i] = o;
}

// ---- Host-side launcher ---------------------------------------------------
extern "C" void kernel_launch(void* const* d_in, const int* in_sizes, int n_in,
                              void* d_out, int out_size, void* d_ws, size_t ws_size,
                              hipStream_t stream) {
  (void)n_in; (void)out_size; (void)ws_size;
  const float* h        = (const float*)d_in[0];
  const float* e        = (const float*)d_in[1];
  const int*   src      = (const int*)d_in[2];
  const int*   dst      = (const int*)d_in[3];
  const float* Wemb_n   = (const float*)d_in[4];
  const float* Wemb_e   = (const float*)d_in[5];
  const float* Wemb_eta = (const float*)d_in[6];
  const float* Uw  = (const float*)d_in[7];  const float* Ub  = (const float*)d_in[8];
  const float* Vw  = (const float*)d_in[9];  const float* Vb  = (const float*)d_in[10];
  const float* W1w = (const float*)d_in[11]; const float* W1b = (const float*)d_in[12];
  const float* W2w = (const float*)d_in[13]; const float* W2b = (const float*)d_in[14];
  const float* W3w = (const float*)d_in[15]; const float* W3b = (const float*)d_in[16];
  const float* h_gamma = (const float*)d_in[17]; const float* h_beta = (const float*)d_in[18];
  const float* e_gamma = (const float*)d_in[19]; const float* e_beta = (const float*)d_in[20];

  const int N = in_sizes[0] / D_DIM;
  const int E = in_sizes[1] / D_DIM;
  const size_t nd = (size_t)N * D_DIM;
  const size_t ed = (size_t)E * D_DIM;

  // Workspace layout
  char* ws = (char*)d_ws;
  _Float16* Wt = (_Float16*)ws;                        // 8 * 128*128 f16
  float* f = (float*)(ws + 8 * 16384 * sizeof(_Float16));
  float* h_emb = f;  f += nd;
  float* Uh    = f;  f += nd;
  float* Vh    = f;  f += nd;
  float* W2h   = f;  f += nd;
  float* W3h   = f;  f += nd;
  float* e_emb = f;  f += ed;
  float* preE  = f;  f += ed;       // later reused as node-side x
  float* num   = f;  f += nd;       // ---- zeroed region start
  float* den   = f;  f += nd;
  float* statsE = f; f += 256;
  float* statsH = f; f += 256;      // ---- zeroed region end
  float* abE = f;    f += 256;
  float* abH = f;    f += 256;
  float* xbuf = preE;               // alias: pre_e dead once e_new is written

  float* out_h = (float*)d_out;     // h_new [N,128]
  float* out_e = out_h + nd;        // e_new [E,128]

  // zero accumulators (num, den, statsE, statsH are contiguous)
  hipMemsetAsync(num, 0, (2 * nd + 512) * sizeof(float), stream);

  // 0) weight transpose+f16 convert. Wt order:
  //    0=W_emb_node 1=Uw 2=Vw 3=W2w 4=W3w 5=W_emb_edge 6=W1w 7=W_emb_eta
  k_wt<<<dim3(64, 8), 256, 0, stream>>>(Wemb_n, Uw, Vw, W2w, W3w, Wemb_e, W1w, Wemb_eta, Wt);

  // 1) node projections (5 GEMMs share A tile)
  const int tilesN = (N + 15) / 16;
  k_node_gemm<<<dim3(tilesN, 5), 32, 0, stream>>>(h, Wt, Ub, Vb, W2b, W3b,
                                                  h_emb, Uh, Vh, W2h, W3h, N);

  // 2) edge GEMMs + gather + BN stats
  const int tilesE = (E + 15) / 16;
  k_edge_gemm<<<(tilesE + 7) / 8, 256, 0, stream>>>(e, Wt, W1b, src, dst, W2h, W3h,
                                                    e_emb, preE, statsE, tilesE);

  // 3) edge BN finalize, 4) e_new
  k_bnfin<<<1, 128, 0, stream>>>(statsE, e_gamma, e_beta, 1.0f / (float)E, abE);
  const int tot4e = (int)(ed / 4);
  k_enew<<<(tot4e + 255) / 256, 256, 0, stream>>>(e_emb, preE, abE, out_e, tot4e);

  // 5) sigma GEMM + scatter-add aggregation
  k_sigma<<<tilesE, 32, 0, stream>>>(out_e, Wt + 7 * 16384, src, dst, Vh, num, den, tilesE);

  // 6) node aggregate + BN stats, 7) node BN finalize, 8) h_new
  k_agg<<<(N + 7) / 8, 256, 0, stream>>>(Uh, num, den, xbuf, statsH, N);
  k_bnfin<<<1, 128, 0, stream>>>(statsH, h_gamma, h_beta, 1.0f / (float)N, abH);
  const int tot4h = (int)(nd / 4);
  k_hnew<<<(tot4h + 255) / 256, 256, 0, stream>>>(h_emb, xbuf, abH, out_h, tot4h);
}
